// DisjGNN_59536836657621
// MI455X (gfx1250) — matrix operand order repaired and checked by reading, not compiled, not run on hardware
//
#include <hip/hip_runtime.h>
#include <stdint.h>

// ---------------------------------------------------------------------------
// CDNA5 (gfx1250) ResNet18 + edge-MLP heads, WMMA everywhere.
//  - Activations: NHWC _Float16; conv = implicit GEMM on
//    v_wmma_f32_16x16x32_f16 (f32 accum), LDS tiles stored fragment-major so
//    each fragment is one 32-byte v16h load (2x ds_load_b128).
//  - Conv weights prepacked to [Cout][(ky*kx)*Cin] f16; BN folded to
//    per-channel scale/bias.
//  - MLP/FC weights prepacked to ZERO-PADDED f16 [Npad][Kpad] so the GEMM
//    kernel has no bounds guards at all: padded output columns compute to
//    exact zeros, which is precisely the K-padding the next layer reads.
//  - All index math is shifts (pow2 dims; KS is a template parameter).
// ---------------------------------------------------------------------------

typedef __attribute__((ext_vector_type(16))) _Float16 v16h;
typedef __attribute__((ext_vector_type(8)))  _Float16 v8h;
typedef __attribute__((ext_vector_type(8)))  float    v8f;

#define BN_EPS 1e-5f

__device__ __forceinline__ v8h v8h_zero() {
  v8h z;
#pragma unroll
  for (int i = 0; i < 8; ++i) z[i] = (_Float16)0.f;
  return z;
}

// ---------------------------------------------------------------------------
// Implicit-GEMM conv + fused BN/residual/ReLU.  NHWC f16 in/out.
//   GEMM: M = Cout, N = Nimg*OH*OW, K = KS*KS*Cin (cin fastest in K).
//   Block: 256 thr = 8 waves; tile 32(Cout) x 64(pix); K chunks of 32.
//   FAST requires Cin % 32 == 0 (every layer except the 7x7 stem).
// Fragment maps (ISA 7.12.2, 16-bit, h = lane>>4):
//   A: e=0..7 <-> k=8h+e ; e=8..15 <-> k=16+8h+(e-8)
//   B: same map as A
// ---------------------------------------------------------------------------
template <int KS, bool FAST>
__global__ void conv_bn_kernel(
    const _Float16* __restrict__ in,     // NHWC
    const _Float16* __restrict__ wp,     // [Cout][K] prepacked f16
    const float* __restrict__ sc,        // folded BN scale  [Cout]
    const float* __restrict__ bi,        // folded BN bias   [Cout]
    const _Float16* residual,            // nullable, NHWC out shape
    _Float16* __restrict__ out,          // NHWC
    int Nimg, int Cin, int lgCin, int H, int W,
    int Cout, int OH, int OW, int osh, int owsh,
    int stride, int pad, int do_relu)
{
  __shared__ __align__(32) _Float16 AsF[2][32][16];   // fragment-major
  __shared__ __align__(32) _Float16 BsF[4][32][16];

  const int K        = KS * KS * Cin;
  const int totalPix = Nimg * OH * OW;
  const int coutBase = blockIdx.x * 32;
  const int pixBase  = blockIdx.y * 64;
  const int tid  = threadIdx.x;
  const int lane = tid & 31;
  const int wv   = tid >> 5;
  const int wm   = wv >> 2;     // 0..1
  const int wn   = wv & 3;      // 0..3
  const int h    = lane >> 4;
  const int l16  = lane & 15;

  v8f acc = {0.f, 0.f, 0.f, 0.f, 0.f, 0.f, 0.f, 0.f};

  for (int k0 = 0; k0 < K; k0 += 32) {
    if (FAST) {
      // ---- A tile: 128 threads copy 8 contiguous halves each ----
      if (tid < 128) {
        int mi = tid & 31, seg = tid >> 5;               // seg 0..3
        int cout = coutBase + mi;
        v8h av = *(const v8h*)(wp + (size_t)cout * K + k0 + 8 * seg);
        *(v8h*)&AsF[mi >> 4][(seg & 1) * 16 + (mi & 15)][(seg >> 1) * 8] = av;
      }
      // ---- B tile: 256 threads, one 16-byte copy each ----
      {
        int p = tid & 63, seg = tid >> 6;                // seg 0..3
        int pix = pixBase + p;
        v8h bv = v8h_zero();
        int rc  = k0 >> lgCin;                           // uniform per chunk
        int rr  = rc / KS, cc = rc % KS;                 // constant division
        int cin0 = k0 & (Cin - 1);
        int n   = pix >> osh;
        int rem = pix & ((1 << osh) - 1);
        int oy  = rem >> owsh, ox = rem & (OW - 1);
        int iy  = oy * stride - pad + rr;
        int ix  = ox * stride - pad + cc;
        if (iy >= 0 && iy < H && ix >= 0 && ix < W) {
          size_t base = ((((size_t)n * H + iy) * W + ix) << lgCin) + cin0 + 8 * seg;
          bv = *(const v8h*)(in + base);
        }
        *(v8h*)&BsF[p >> 4][(seg & 1) * 16 + (p & 15)][(seg >> 1) * 8] = bv;
      }
    } else {
      // ---- generic scalar staging (7x7 stem: Cin=4, K=196) ----
      for (int i = tid; i < 32 * 32; i += 256) {
        int mi = i >> 5, kk = i & 31;
        int cout = coutBase + mi, k = k0 + kk;
        _Float16 val = (_Float16)0.f;
        if (cout < Cout && k < K) val = wp[(size_t)cout * K + k];
        int hh = (kk >> 3) & 1;
        int e  = (kk < 16) ? (kk - 8 * hh) : (kk - 8 - 8 * hh);
        AsF[mi >> 4][hh * 16 + (mi & 15)][e] = val;
      }
      for (int i = tid; i < 32 * 64; i += 256) {
        int kk = i >> 6, p = i & 63;
        int pix = pixBase + p, k = k0 + kk;
        _Float16 val = (_Float16)0.f;
        if (pix < totalPix && k < K) {
          int cin = k & (Cin - 1);
          int rc  = k >> lgCin;
          int rr  = rc / KS, cc = rc % KS;
          int n   = pix >> osh;
          int rem = pix & ((1 << osh) - 1);
          int oy  = rem >> owsh, ox = rem & (OW - 1);
          int iy  = oy * stride - pad + rr;
          int ix  = ox * stride - pad + cc;
          if (iy >= 0 && iy < H && ix >= 0 && ix < W)
            val = in[((((size_t)n * H + iy) * W + ix) << lgCin) + cin];
        }
        int hh = (kk >> 3) & 1, e = ((kk >> 4) << 3) + (kk & 7);
        BsF[p >> 4][hh * 16 + (p & 15)][e] = val;
      }
    }
    __syncthreads();

    v16h a  = *(const v16h*)&AsF[wm][lane][0];   // 2x ds_load_b128
    v16h bf = *(const v16h*)&BsF[wn][lane][0];
    acc = __builtin_amdgcn_wmma_f32_16x16x32_f16(
        false, a, false, bf, (short)0, acc, false, false);
    __syncthreads();
  }

  // ---- epilogue: BN + residual + ReLU; one contiguous v8h store ----
  // (totalPix multiple of 64 and Cout of 32 for every layer -> no guards)
  {
    int col = pixBase + 16 * wn + l16;
    int n   = col >> osh;
    int rem = col & ((1 << osh) - 1);
    int oy  = rem >> owsh, ox = rem & (OW - 1);
    int c0  = coutBase + 16 * wm + 8 * h;        // 8 consecutive channels
    size_t obase = ((((size_t)n * OH + oy) * OW + ox) * (size_t)Cout) + c0;
    v8h o;
    if (residual) {
      v8h rv = *(const v8h*)(residual + obase);
#pragma unroll
      for (int r = 0; r < 8; ++r) {
        float y = acc[r] * sc[c0 + r] + bi[c0 + r] + (float)rv[r];
        if (do_relu) y = y > 0.f ? y : 0.f;
        o[r] = (_Float16)y;
      }
    } else {
#pragma unroll
      for (int r = 0; r < 8; ++r) {
        float y = acc[r] * sc[c0 + r] + bi[c0 + r];
        if (do_relu) y = y > 0.f ? y : 0.f;
        o[r] = (_Float16)y;
      }
    }
    *(v8h*)(out + obase) = o;
  }
}

// ---------------------------------------------------------------------------
// One-time prep kernels
// ---------------------------------------------------------------------------
// OIHW f32 -> [Cout][(ky*kx)*Cin] f16 (cin fastest)
__global__ void wpack_kernel(const float* __restrict__ w, _Float16* __restrict__ wp,
                             int Cout, int Cin, int ks)
{
  int i = blockIdx.x * blockDim.x + threadIdx.x;
  int KK = ks * ks;
  int total = Cout * Cin * KK;
  if (i >= total) return;
  int cin = i % Cin; int t = i / Cin;
  int rc = t % KK;   int cout = t / KK;
  int rr = rc / ks,  cc = rc % ks;
  wp[i] = (_Float16)w[(((size_t)cout * Cin + cin) * ks + rr) * ks + cc];
}

__global__ void bnfold_kernel(const float* g, const float* b, const float* m,
                              const float* v, float* sc, float* bi, int C)
{
  int i = blockIdx.x * blockDim.x + threadIdx.x;
  if (i >= C) return;
  float s = g[i] * rsqrtf(v[i] + BN_EPS);
  sc[i] = s;
  bi[i] = b[i] - m[i] * s;
}

// MLP/FC weights: f32 [Nout][Kt] -> zero-padded f16 [Npad][Kpad] (+padded bias)
__global__ void mlp_pack_kernel(const float* __restrict__ w, const float* __restrict__ b,
                                int Nout, int Kt, int lgKpad,
                                _Float16* __restrict__ wp, float* __restrict__ bp,
                                int total)
{
  int i = blockIdx.x * blockDim.x + threadIdx.x;
  if (i >= total) return;
  int Kpad = 1 << lgKpad;
  int k = i & (Kpad - 1);
  int n = i >> lgKpad;
  float v = (n < Nout && k < Kt) ? w[(size_t)n * Kt + k] : 0.f;
  wp[i] = (_Float16)v;
  if (k == 0) bp[n] = (n < Nout) ? b[n] : 0.f;
}

// NCHW f32 -> NHWC f16
__global__ void nchw2nhwc_kernel(const float* __restrict__ x, _Float16* __restrict__ y,
                                 int lgC, int lgW, int lgH, int total)
{
  int i = blockIdx.x * blockDim.x + threadIdx.x;
  if (i >= total) return;
  int C = 1 << lgC, W = 1 << lgW, H = 1 << lgH;
  int c = i & (C - 1); int t = i >> lgC;
  int ix = t & (W - 1); t >>= lgW;
  int iy = t & (H - 1); int n = t >> lgH;
  y[i] = (_Float16)x[(((size_t)n * C + c) * H + iy) * W + ix];
}

// 3x3 s2 p1 maxpool, NHWC f16
__global__ void maxpool_kernel(const _Float16* __restrict__ in, _Float16* __restrict__ out,
                               int lgC, int H, int W, int OH, int OW,
                               int lgOW, int lgOH, int total)
{
  int i = blockIdx.x * blockDim.x + threadIdx.x;
  if (i >= total) return;
  int C = 1 << lgC;
  int c = i & (C - 1); int t = i >> lgC;
  int ox = t & (OW - 1); t >>= lgOW;
  int oy = t & (OH - 1); int n = t >> lgOH;
  float m = -3.4e38f;
  for (int r = 0; r < 3; ++r)
    for (int cc = 0; cc < 3; ++cc) {
      int iy = oy * 2 - 1 + r, ix = ox * 2 - 1 + cc;
      if (iy >= 0 && iy < H && ix >= 0 && ix < W) {
        float v = (float)in[((((size_t)n * H + iy) * W + ix)) * C + c];
        m = v > m ? v : m;
      }
    }
  out[i] = (_Float16)m;
}

// global average pool, NHWC f16 -> [N][C] f16
__global__ void avgpool_kernel(const _Float16* __restrict__ in, _Float16* __restrict__ out,
                               int lgC, int HW, int total)
{
  int i = blockIdx.x * blockDim.x + threadIdx.x;
  if (i >= total) return;
  int C = 1 << lgC;
  int c = i & (C - 1); int n = i >> lgC;
  size_t base = ((size_t)n * HW) << lgC;
  float s = 0.f;
  for (int j = 0; j < HW; ++j) s += (float)in[base + ((size_t)j << lgC) + c];
  out[i] = (_Float16)(s / (float)HW);
}

// feats[e] = concat(out_x[src[e]], out_x[dst[e]]); F=64
__global__ void gather_edge_kernel(const _Float16* __restrict__ outx,
                                   const int* __restrict__ ei,
                                   int E, int F, int lg2F,
                                   _Float16* __restrict__ feats)
{
  int i = blockIdx.x * blockDim.x + threadIdx.x;
  if (i >= (E << lg2F)) return;
  int j = i & ((1 << lg2F) - 1);
  int e = i >> lg2F;
  int node = (j < F) ? ei[e] : ei[E + e];
  int c    = (j < F) ? j : j - F;
  feats[i] = outx[node * F + c];
}

// ---------------------------------------------------------------------------
// Guard-free WMMA GEMM: out = act(A[f16, MxKpad] * Wp[f16, NpadxKpad]^T + Bp)
//   - A zero-padded to Kpad (previous layer writes full tiles).
//   - Wp/Bp zero-padded, so tile columns >= true width compute exact zeros
//     (which IS the next layer's K padding).
//   - Row clamp for loads; stores guarded by row < M only.
// Block 256 thr = 8 waves; tile 128(M) x 16(N); K chunks of 32.
// ---------------------------------------------------------------------------
__global__ void gemm_kernel(
    const _Float16* __restrict__ A, int lda, int M, int Kpad,
    const _Float16* __restrict__ Wp,          // [Npad][Kpad] padded f16
    const float* __restrict__ Bp,             // [Npad] padded bias
    _Float16* outh, int ldo,                  // nullable f16 output
    float* outf, int ldof, int Noutf,         // nullable f32 output, col<Noutf
    int do_relu)
{
  const int lane = threadIdx.x & 31;
  const int wv   = threadIdx.x >> 5;
  const int rowBase = blockIdx.x * 128 + 16 * wv;
  const int colBase = blockIdx.y * 16;
  const int h    = lane >> 4;
  const int l16  = lane & 15;
  int arow = rowBase + l16; arow = arow < M ? arow : M - 1;   // clamp
  const int bcol = colBase + l16;
  const _Float16* ap = A  + (size_t)arow * lda;
  const _Float16* wp = Wp + (size_t)bcol * Kpad;

  v8f acc = {0.f, 0.f, 0.f, 0.f, 0.f, 0.f, 0.f, 0.f};

  for (int k0 = 0; k0 < Kpad; k0 += 32) {
    if (k0 + 32 < Kpad)
      __builtin_prefetch(ap + k0 + 32, 0, 1);     // global_prefetch_b8
    v8h lo = *(const v8h*)(ap + k0 + 16 * h);     // A: e=0..7  (k=16h..)
    v8h hi = *(const v8h*)(ap + k0 + 16 * h + 8); // A: e=8..15 (k=16h+8..)
    v16h a = __builtin_shufflevector(lo, hi, 0, 1, 2, 3, 4, 5, 6, 7,
                                     8, 9, 10, 11, 12, 13, 14, 15);
    v16h b = *(const v16h*)(wp + k0 + 16 * h);    // B: 16 contiguous halves
    acc = __builtin_amdgcn_wmma_f32_16x16x32_f16(
        false, a, false, b, (short)0, acc, false, false);
  }

  const int col = colBase + l16;
  const float bs = Bp[col];
#pragma unroll
  for (int r = 0; r < 8; ++r) {
    int row = rowBase + 8 * h + r;
    if (row < M) {
      float y = acc[r] + bs;
      if (do_relu) y = y > 0.f ? y : 0.f;
      if (outh) outh[(size_t)row * ldo + col] = (_Float16)y;
      if (outf && col < Noutf) outf[(size_t)row * ldof + col] = y;
    }
  }
}

// ---------------------------------------------------------------------------
// Host orchestration
// ---------------------------------------------------------------------------
namespace {
struct BNP { const float *g, *b, *m, *v; };
struct Packed { const _Float16* wp; const float* sc; const float* bi; };
struct MPack { const _Float16* wp; const float* bp; int Kpad; int Npad; };
inline int cdiv(int a, int b) { return (a + b - 1) / b; }
inline int ilog2(int x) { int s = 0; while ((1 << s) < x) ++s; return s; }
inline int rup32(int x) { return (x + 31) & ~31; }
}

extern "C" void kernel_launch(void* const* d_in, const int* in_sizes, int n_in,
                              void* d_out, int out_size, void* d_ws, size_t ws_size,
                              hipStream_t stream) {
  (void)in_sizes; (void)n_in; (void)out_size; (void)ws_size;
  const int Nimg = 64, E = 4096;

  // ---- walk flattened inputs in setup_inputs() insertion order ----
  int cur = 0;
  auto nf = [&]() { return (const float*)d_in[cur++]; };
  const float* x = nf();
  const int* edges = (const int*)d_in[cur++];   // edge_index [2,E]
  auto nbn = [&]() { BNP p; p.g = nf(); p.b = nf(); p.m = nf(); p.v = nf(); return p; };

  const float* conv1_w = nf();
  BNP bn1 = nbn();

  const int cout_[8]   = {64, 64, 128, 128, 256, 256, 512, 512};
  const int stride_[8] = {1, 1, 2, 1, 2, 1, 2, 1};
  const float* bw1[8]; BNP bbn1[8];
  const float* bw2[8]; BNP bbn2[8];
  const float* bwd[8]; BNP bbnd[8];
  int cin = 64;
  for (int i = 0; i < 8; ++i) {
    bw1[i] = nf(); bbn1[i] = nbn();
    bw2[i] = nf(); bbn2[i] = nbn();
    bool ds = (stride_[i] != 1) || (cin != cout_[i]);
    if (ds) { bwd[i] = nf(); bbnd[i] = nbn(); } else { bwd[i] = nullptr; }
    cin = cout_[i];
  }
  const float* fc_w = nf();
  const float* fc_b = nf();
  const float* mw[3][6]; const float* mb[3][6];
  for (int hd = 0; hd < 3; ++hd)
    for (int l = 0; l < 6; ++l) { mw[hd][l] = nf(); mb[hd][l] = nf(); }

  // ---- workspace bump allocator ----
  char* wsp = (char*)d_ws;
  size_t off = 0;
  auto walloc = [&](size_t bytes) -> void* {
    void* p = wsp + off;
    off = (off + bytes + 255) & ~(size_t)255;
    return p;
  };
  const size_t BIG = (size_t)Nimg * 64 * 128 * 128 * sizeof(_Float16);
  _Float16* xh = (_Float16*)walloc((size_t)Nimg * 4 * 256 * 256 * sizeof(_Float16));
  _Float16* P0 = (_Float16*)walloc(BIG);
  _Float16* P1 = (_Float16*)walloc(BIG);
  _Float16* T  = (_Float16*)walloc((size_t)Nimg * 128 * 32 * 32 * sizeof(_Float16));
  _Float16* feats  = (_Float16*)walloc((size_t)Nimg * 512 * sizeof(_Float16));
  _Float16* outx_h = (_Float16*)walloc((size_t)Nimg * 64 * sizeof(_Float16));
  _Float16* Efeat  = (_Float16*)walloc((size_t)E * 128 * sizeof(_Float16));
  _Float16* Ep0    = (_Float16*)walloc((size_t)E * 64 * sizeof(_Float16));
  _Float16* Ep1    = (_Float16*)walloc((size_t)E * 64 * sizeof(_Float16));

  // ---- prep: conv weights -> f16 (cin fastest), fold BN ----
  auto prep = [&](const float* w, BNP bn, int Cout, int Cin, int ks) -> Packed {
    int K = ks * ks * Cin;
    _Float16* wp = (_Float16*)walloc((size_t)Cout * K * sizeof(_Float16));
    float* sc = (float*)walloc(Cout * sizeof(float));
    float* bi = (float*)walloc(Cout * sizeof(float));
    int total = Cout * K;
    wpack_kernel<<<cdiv(total, 256), 256, 0, stream>>>(w, wp, Cout, Cin, ks);
    bnfold_kernel<<<cdiv(Cout, 256), 256, 0, stream>>>(bn.g, bn.b, bn.m, bn.v,
                                                       sc, bi, Cout);
    Packed p; p.wp = wp; p.sc = sc; p.bi = bi; return p;
  };
  // ---- prep: MLP/FC weights -> zero-padded f16 [Npad][Kpad] ----
  auto prep_mlp = [&](const float* w, const float* b, int Nout, int Kt,
                      int Npad) -> MPack {
    int Kpad = rup32(Kt);
    _Float16* wp = (_Float16*)walloc((size_t)Npad * Kpad * sizeof(_Float16));
    float* bp = (float*)walloc(Npad * sizeof(float));
    int total = Npad * Kpad;
    mlp_pack_kernel<<<cdiv(total, 256), 256, 0, stream>>>(
        w, b, Nout, Kt, ilog2(Kpad), wp, bp, total);
    MPack p; p.wp = wp; p.bp = bp; p.Kpad = Kpad; p.Npad = Npad; return p;
  };

  Packed pk1 = prep(conv1_w, bn1, 64, 4, 7);
  Packed pkb1[8], pkb2[8], pkbd[8];
  cin = 64;
  for (int i = 0; i < 8; ++i) {
    pkb1[i] = prep(bw1[i], bbn1[i], cout_[i], cin, 3);
    pkb2[i] = prep(bw2[i], bbn2[i], cout_[i], cout_[i], 3);
    if (bwd[i]) pkbd[i] = prep(bwd[i], bbnd[i], cout_[i], cin, 1);
    cin = cout_[i];
  }
  MPack fcp = prep_mlp(fc_w, fc_b, 64, 512, 64);
  // widths: 128 -> 64 -> 32 -> 16 -> 8 -> 4 -> 1
  const int widths[7] = {128, 64, 32, 16, 8, 4, 1};
  MPack mpk[3][6];
  for (int hd = 0; hd < 3; ++hd)
    for (int l = 0; l < 6; ++l) {
      int Npad = (l == 5) ? 16 : ((widths[l + 1] < 32) ? 32 : widths[l + 1]);
      mpk[hd][l] = prep_mlp(mw[hd][l], mb[hd][l], widths[l + 1], widths[l], Npad);
    }

  auto conv = [&](const _Float16* in, Packed pk, const _Float16* res, _Float16* out,
                  int Cin, int H, int Cout, int OH, int ks, int st, int pad, int relu) {
    dim3 grid(Cout / 32, cdiv(Nimg * OH * OH, 64));
    int osh = ilog2(OH * OH), owsh = ilog2(OH), lc = ilog2(Cin);
    if (ks == 7)
      conv_bn_kernel<7, false><<<grid, 256, 0, stream>>>(
          in, pk.wp, pk.sc, pk.bi, res, out, Nimg, Cin, lc, H, H,
          Cout, OH, OH, osh, owsh, st, pad, relu);
    else if (ks == 3)
      conv_bn_kernel<3, true><<<grid, 256, 0, stream>>>(
          in, pk.wp, pk.sc, pk.bi, res, out, Nimg, Cin, lc, H, H,
          Cout, OH, OH, osh, owsh, st, pad, relu);
    else
      conv_bn_kernel<1, true><<<grid, 256, 0, stream>>>(
          in, pk.wp, pk.sc, pk.bi, res, out, Nimg, Cin, lc, H, H,
          Cout, OH, OH, osh, owsh, st, pad, relu);
  };
  auto gemm = [&](const _Float16* A, int lda, int M, MPack mp,
                  _Float16* oh, int ldo, float* of, int ldof, int Noutf, int relu) {
    dim3 grid(cdiv(M, 128), mp.Npad / 16);
    gemm_kernel<<<grid, 256, 0, stream>>>(
        A, lda, M, mp.Kpad, mp.wp, mp.bp, oh, ldo, of, ldof, Noutf, relu);
  };

  // ---- stem ----
  {
    int total = Nimg * 4 * 256 * 256;
    nchw2nhwc_kernel<<<cdiv(total, 256), 256, 0, stream>>>(x, xh, 2, 8, 8, total);
  }
  conv(xh, pk1, nullptr, P0, 4, 256, 64, 128, 7, 2, 3, 1);
  {
    int total = Nimg * 64 * 64 * 64;
    maxpool_kernel<<<cdiv(total, 256), 256, 0, stream>>>(
        P0, P1, 6, 128, 128, 64, 64, 6, 6, total);
  }

  // ---- residual blocks (NHWC everywhere) ----
  _Float16* act = P1;
  _Float16* tmp = P0;
  cin = 64; int hs = 64;
  for (int i = 0; i < 8; ++i) {
    int co = cout_[i], st = stride_[i], oh = hs / st;
    conv(act, pkb1[i], nullptr, tmp, cin, hs, co, oh, 3, st, 1, 1);
    const _Float16* res = act;
    if (bwd[i]) {
      conv(act, pkbd[i], nullptr, T, cin, hs, co, oh, 1, st, 0, 0);
      res = T;
    }
    conv(tmp, pkb2[i], res, act, co, oh, co, oh, 3, 1, 1, 1);
    cin = co; hs = oh;
  }

  // ---- avg pool -> FC; out_x f32 at d_out[4096..8191] ----
  {
    int total = Nimg * 512;
    avgpool_kernel<<<cdiv(total, 256), 256, 0, stream>>>(act, feats, 9, hs * hs, total);
  }
  float* out_f = (float*)d_out;
  gemm(feats, 512, Nimg, fcp, outx_h, 64, out_f + 4096, 64, 64, 0);

  // ---- edge gather + 3 MLP heads ----
  gather_edge_kernel<<<cdiv(E * 128, 256), 256, 0, stream>>>(
      outx_h, edges, E, 64, 7, Efeat);
  const int head_off[3] = {0, 8192, 12288};   // sim, pos, rot
  for (int hd = 0; hd < 3; ++hd) {
    gemm(Efeat, 128, E, mpk[hd][0], Ep0, 64, nullptr, 0, 0, 1);
    _Float16* a = Ep0; _Float16* b = Ep1;
    for (int l = 1; l < 5; ++l) {
      gemm(a, 64, E, mpk[hd][l], b, 64, nullptr, 0, 0, 1);
      _Float16* t2 = a; a = b; b = t2;
    }
    gemm(a, 64, E, mpk[hd][5], nullptr, 0, out_f + head_off[hd], 1, 1, 0);
  }
}
